// EntropyMoE_38354057953725
// MI455X (gfx1250) — matrix operand order repaired and checked by reading, not compiled, not run on hardware
//
#include <hip/hip_runtime.h>
#include <math.h>

// ---------------- CDNA5 (gfx1250) types ----------------
typedef __attribute__((ext_vector_type(16))) __bf16 v16bf;
typedef __attribute__((ext_vector_type(8)))  float  v8f;

union FragBF {
    v16bf v;
    uint4 u[2];
};

__device__ __forceinline__ unsigned short f2bf(float f) {
    // round-to-nearest-even fp32 -> bf16
    unsigned int u = __float_as_uint(f);
    u = u + 0x7FFFu + ((u >> 16) & 1u);
    return (unsigned short)(u >> 16);
}
__device__ __forceinline__ unsigned int f2bf_pk(float lo, float hi) {
    return (unsigned int)f2bf(lo) | ((unsigned int)f2bf(hi) << 16);
}
__device__ __forceinline__ float bf2f(unsigned short h) {
    return __uint_as_float(((unsigned int)h) << 16);
}
__device__ __forceinline__ float gelu_exact(float v) {
    return 0.5f * v * (1.0f + erff(v * 0.70710678118654752f));
}

// ---- CDNA5 async global->LDS copy (ASYNCcnt path) ----
__device__ __forceinline__ void async_copy_b128x2(unsigned lds_off, const void* gaddr) {
    unsigned long long ga = (unsigned long long)(uintptr_t)gaddr;
    // IOFFSET is added to BOTH the LDS and global address (ISA 10.7 pseudocode),
    // so one VGPR setup serves two 16B chunks.
    asm volatile("global_load_async_to_lds_b128 %0, %1, off"
                 :: "v"(lds_off), "v"(ga) : "memory");
    asm volatile("global_load_async_to_lds_b128 %0, %1, off offset:16"
                 :: "v"(lds_off), "v"(ga) : "memory");
}
__device__ __forceinline__ void wait_async_le2() {
#if __has_builtin(__builtin_amdgcn_s_wait_asynccnt)
    __builtin_amdgcn_s_wait_asynccnt(2);
#else
    asm volatile("s_wait_asynccnt 0x2" ::: "memory");
#endif
}
__device__ __forceinline__ void wait_async_le0() {
#if __has_builtin(__builtin_amdgcn_s_wait_asynccnt)
    __builtin_amdgcn_s_wait_asynccnt(0);
#else
    asm volatile("s_wait_asynccnt 0x0" ::: "memory");
#endif
}

// =====================================================================
// Generic tiled bf16 WMMA GEMM:  C_tile = A(bf16,[MxK]) * B(f32,[KxN]->bf16)
//   mode 0: out_bf[row*N+col] = bf16( gelu( acc + bias[col] ) )
//   mode 1: out_f [row*N+col] += scale[row*scaleStride] * ( acc + bias[col] )
// Block = 256 threads = 8 waves (wave32).  Tile 128x128, K-step 32,
// K-loop unrolled x2 with ping-pong B register sets (no reg copies,
// so the next-tile global loads can ride under the WMMA block).
// A tiles: double-buffered LDS filled by global_load_async_to_lds_b128,
// issued one K-step ahead; s_wait_asynccnt 2 proves the current tile
// landed (async loads complete in order).
// =====================================================================
#define TM 128
#define TN 128
#define TK 32
#define LDS_STRIDE 40   // halves per LDS row: 32 data + 8 pad (80B, keeps 16B align)

__global__ __launch_bounds__(256) void gemm_bf16_wmma(
    const unsigned short* __restrict__ A, int lda,
    const float* __restrict__ Bsrc, int ldb,
    int N, int K,
    const float* __restrict__ bias,          // len N, or nullptr
    int mode,
    unsigned short* __restrict__ outBf,      // mode 0
    float* __restrict__ outF,                // mode 1
    const float* __restrict__ scale, int scaleStride)
{
    __shared__ __align__(16) unsigned short Abuf[2][TM * LDS_STRIDE];
    __shared__ __align__(16) unsigned short Bs[TN * LDS_STRIDE];

    const int tid  = threadIdx.x;
    const int lane = tid & 31;
    const int w    = tid >> 5;
    const int wm   = w & 3;           // 4 waves along M
    const int wn   = w >> 2;          // 2 waves along N
    const int m0   = blockIdx.y * TM;
    const int nb   = blockIdx.x * TN;

    v8f acc[2][4] = {};

    // A staging: each thread owns one 32B chunk (16 bf16) of the 128x32 tile
    const int ar = tid >> 1;          // row 0..127
    const int as = (tid & 1) * 16;    // half-offset 0 or 16
    // B staging: 32 thread-columns x 8 even-k rows; pairs (k0,k0+1), 2 passes
    const int bn  = (tid & 31) * 4;   // n offset 0..124
    const int bkp = (tid >> 5) * 2;   // even k base 0..14

    unsigned ldsA[2];
    ldsA[0] = (unsigned)(uintptr_t)&Abuf[0][ar * LDS_STRIDE + as];
    ldsA[1] = (unsigned)(uintptr_t)&Abuf[1][ar * LDS_STRIDE + as];
    const unsigned short* Arow = A + (size_t)(m0 + ar) * lda + as;

    const int k0a = (lane >> 4) * 8;    // A: half-wave K split {0..7,16..23}/{8..15,24..31}
    const int k0b = (lane >> 4) * 8;    // B: half-wave K split {0..7,16..23}/{8..15,24..31}

    float4 b0A[2], b0B[2], b1A[2], b1B[2];   // ping-pong B register sets

#define LOADB(KB, DA, DB)                                                        \
    {                                                                            \
        _Pragma("unroll")                                                        \
        for (int p = 0; p < 2; ++p) {                                            \
            int k0 = p * 16 + bkp;                                               \
            DA[p] = *(const float4*)(Bsrc + (size_t)((KB) + k0    ) * ldb + nb + bn); \
            DB[p] = *(const float4*)(Bsrc + (size_t)((KB) + k0 + 1) * ldb + nb + bn); \
        }                                                                        \
    }

#define KSTEP(BUF, KB, CA, CB, NA, NB)                                           \
    {                                                                            \
        const bool hasNext = ((KB) + TK) < K;                                    \
        __syncthreads(); /* prev compute done: Bs free, Abuf[BUF^1] free */      \
        if (hasNext) async_copy_b128x2(ldsA[(BUF) ^ 1], Arow + (KB) + TK);       \
        _Pragma("unroll")                                                        \
        for (int p = 0; p < 2; ++p) {                                            \
            int k0 = p * 16 + bkp;                                               \
            const float* fA = (const float*)&CA[p];                              \
            const float* fB = (const float*)&CB[p];                              \
            _Pragma("unroll")                                                    \
            for (int j = 0; j < 4; ++j)                                          \
                *(unsigned int*)&Bs[(bn + j) * LDS_STRIDE + k0] = f2bf_pk(fA[j], fB[j]); \
        }                                                                        \
        if (hasNext) LOADB((KB) + TK, NA, NB);  /* overlaps with WMMAs below */  \
        if (hasNext) wait_async_le2(); else wait_async_le0();                    \
        __syncthreads(); /* all waves: current A + B tiles resident */           \
        const unsigned short* Ab = Abuf[BUF];                                    \
        FragBF afr[2], bfr[4];                                                   \
        _Pragma("unroll")                                                        \
        for (int mt = 0; mt < 2; ++mt) {                                         \
            int r = wm * 32 + mt * 16 + (lane & 15);                             \
            afr[mt].u[0] = *(const uint4*)(&Ab[r * LDS_STRIDE + k0a]);           \
            afr[mt].u[1] = *(const uint4*)(&Ab[r * LDS_STRIDE + k0a + 16]);      \
        }                                                                        \
        _Pragma("unroll")                                                        \
        for (int nt = 0; nt < 4; ++nt) {                                         \
            int n = wn * 64 + nt * 16 + (lane & 15);                             \
            bfr[nt].u[0] = *(const uint4*)(&Bs[n * LDS_STRIDE + k0b]);           \
            bfr[nt].u[1] = *(const uint4*)(&Bs[n * LDS_STRIDE + k0b + 16]);      \
        }                                                                        \
        _Pragma("unroll")                                                        \
        for (int mt = 0; mt < 2; ++mt)                                           \
            _Pragma("unroll")                                                    \
            for (int nt = 0; nt < 4; ++nt)                                       \
                acc[mt][nt] = __builtin_amdgcn_wmma_f32_16x16x32_bf16(           \
                    false, afr[mt].v, false, bfr[nt].v,                          \
                    (short)0, acc[mt][nt], false, false);                        \
    }

    // ---- prologue: tile 0 in flight ----
    async_copy_b128x2(ldsA[0], Arow);
    LOADB(0, b0A, b0B);

    // K/TK is even for all call sites (64 or 256 steps)
    for (int kb = 0; kb < K; kb += 2 * TK) {
        KSTEP(0, kb,      b0A, b0B, b1A, b1B);
        KSTEP(1, kb + TK, b1A, b1B, b0A, b0B);
    }
#undef KSTEP
#undef LOADB

    // ---- epilogue (C layout: VGPR v -> row v / v+8 per half-wave, col = lane&15) ----
    #pragma unroll
    for (int mt = 0; mt < 2; ++mt) {
        #pragma unroll
        for (int nt = 0; nt < 4; ++nt) {
            int col   = nb + wn * 64 + nt * 16 + (lane & 15);
            int rbase = m0 + wm * 32 + mt * 16 + 8 * (lane >> 4);
            float bv  = bias ? bias[col] : 0.0f;
            #pragma unroll
            for (int v = 0; v < 8; ++v) {
                int row = rbase + v;
                float val = acc[mt][nt][v] + bv;
                if (mode == 0) {
                    outBf[(size_t)row * N + col] = f2bf(gelu_exact(val));
                } else {
                    size_t o = (size_t)row * N + col;
                    outF[o] += scale[(size_t)row * scaleStride] * val;
                }
            }
        }
    }
}

// =====================================================================
// Small helper kernels
// =====================================================================
__global__ void k_cvt_bf16(const float* __restrict__ in, unsigned short* __restrict__ out, int n) {
    int i = blockIdx.x * 256 + threadIdx.x;
    if (i < n) out[i] = f2bf(in[i]);
}

__global__ void k_zero_flags(unsigned int* flags) {
    if (threadIdx.x < 2) flags[threadIdx.x] = 0u;
}

__global__ void k_gelu_vec(const float* __restrict__ in, float* __restrict__ out, int n) {
    int i = blockIdx.x * 256 + threadIdx.x;
    if (i < n) out[i] = gelu_exact(in[i]);
}

// logits = Hr @ rw2 ; softmax ; top-2 (lowest-index tiebreak) ; active flags
__global__ __launch_bounds__(256) void k_router_topk(
    const unsigned short* __restrict__ Hr,   // [M x 1024] bf16 = gelu(x@rw1)
    const float* __restrict__ rw2,           // [1024 x 8]
    float* __restrict__ wgt,                 // [M x 2]
    int*   __restrict__ idxA,                // [M x 2]
    unsigned int* __restrict__ flags)        // [2] bitmasks over experts
{
    __shared__ float rw2s[1024 * 8];
    for (int i = threadIdx.x; i < 1024 * 8; i += 256) rw2s[i] = rw2[i];
    __syncthreads();

    int t = blockIdx.x * 256 + threadIdx.x;
    const unsigned short* hrow = Hr + (size_t)t * 1024;
    float logit[8] = {};
    for (int i = 0; i < 1024; ++i) {
        float h = bf2f(hrow[i]);
        #pragma unroll
        for (int e = 0; e < 8; ++e) logit[e] += h * rw2s[i * 8 + e];
    }
    float mx = logit[0];
    #pragma unroll
    for (int e = 1; e < 8; ++e) mx = fmaxf(mx, logit[e]);
    float p[8]; float s = 0.0f;
    #pragma unroll
    for (int e = 0; e < 8; ++e) { p[e] = expf(logit[e] - mx); s += p[e]; }
    float inv = 1.0f / s;
    #pragma unroll
    for (int e = 0; e < 8; ++e) p[e] *= inv;

    int i0 = 0; float m0 = p[0];
    #pragma unroll
    for (int e = 1; e < 8; ++e) if (p[e] > m0) { m0 = p[e]; i0 = e; }
    int i1 = (i0 == 0) ? 1 : 0; float m1 = p[i1];
    #pragma unroll
    for (int e = 0; e < 8; ++e) if (e != i0 && p[e] > m1) { m1 = p[e]; i1 = e; }

    wgt[t * 2 + 0] = m0;  wgt[t * 2 + 1] = m1;
    idxA[t * 2 + 0] = i0; idxA[t * 2 + 1] = i1;
    atomicOr(&flags[0], 1u << i0);
    atomicOr(&flags[1], 1u << i1);
}

// coeff[t,e] = sum_k wgt*[idx==e];  gamma[t,e] = sum_k wgt*active[e,k] - coeff
__global__ void k_coeff_gamma(const float* __restrict__ wgt, const int* __restrict__ idxA,
                              const unsigned int* __restrict__ flags,
                              float* __restrict__ coeff, float* __restrict__ gam)
{
    int t = blockIdx.x * 256 + threadIdx.x;
    float w0 = wgt[t * 2], w1 = wgt[t * 2 + 1];
    int   i0 = idxA[t * 2], i1 = idxA[t * 2 + 1];
    unsigned f0 = flags[0], f1 = flags[1];
    #pragma unroll
    for (int e = 0; e < 8; ++e) {
        float ce = (e == i0 ? w0 : 0.0f) + (e == i1 ? w1 : 0.0f);
        float sc = w0 * (float)((f0 >> e) & 1u) + w1 * (float)((f1 >> e) & 1u);
        coeff[t * 8 + e] = ce;
        gam[t * 8 + e]   = sc - ce;
    }
}

// Ce[e,d] = gelu(b1[e]) @ w2[e][:,d] + b2[e,d]
__global__ void k_bias_leak(const float* __restrict__ gb1, const float* __restrict__ w2,
                            const float* __restrict__ b2, float* __restrict__ Ce,
                            int D, int I)
{
    int e = blockIdx.y;
    int d = blockIdx.x * 256 + threadIdx.x;
    const float* w2e = w2 + (size_t)e * I * D;
    const float* g   = gb1 + (size_t)e * I;
    float c = b2[e * D + d];
    #pragma unroll 4
    for (int i = 0; i < I; ++i) c = fmaf(g[i], w2e[(size_t)i * D + d], c);
    Ce[e * D + d] = c;
}

// out = Gamma @ Ce   (rank-8 bias-leak correction; initializes d_out)
__global__ void k_init_out(const float* __restrict__ gam, const float* __restrict__ Ce,
                           float* __restrict__ out, int n)
{
    int i = blockIdx.x * 256 + threadIdx.x;
    if (i >= n) return;
    int row = i >> 11;        // D = 2048
    int d   = i & 2047;
    float a = 0.0f;
    #pragma unroll
    for (int e = 0; e < 8; ++e) a = fmaf(gam[row * 8 + e], Ce[e * 2048 + d], a);
    out[i] = a;
}

// =====================================================================
// Launcher
// =====================================================================
extern "C" void kernel_launch(void* const* d_in, const int* in_sizes, int n_in,
                              void* d_out, int out_size, void* d_ws, size_t ws_size,
                              hipStream_t stream)
{
    const float* x   = (const float*)d_in[0];   // [B,T,D]
    const float* rw1 = (const float*)d_in[1];   // [D, D/2]
    const float* rw2 = (const float*)d_in[2];   // [D/2, E]
    const float* w1  = (const float*)d_in[3];   // [E, D, I]
    const float* b1  = (const float*)d_in[4];   // [E, I]
    const float* w2  = (const float*)d_in[5];   // [E, I, D]
    const float* b2  = (const float*)d_in[6];   // [E, D]
    float* out = (float*)d_out;

    const int D = 2048, I = 8192, E = 8;
    const int M = 4096;                  // B*T tokens
    const int H2 = D / 2;                // 1024 router hidden

    // ---- workspace layout (~85 MB) ----
    char* p = (char*)d_ws;
    auto take = [&](size_t bytes) { char* r = p; p += (bytes + 255) & ~(size_t)255; return r; };
    unsigned short* xb   = (unsigned short*)take((size_t)M * D * 2);   // x in bf16 (L2-resident)
    unsigned short* Hbig = (unsigned short*)take((size_t)M * I * 2);   // Hr (M x 1024) then H (M x 8192)
    float* wgt   = (float*)take((size_t)M * 2 * 4);
    int*   idxA  = (int*)  take((size_t)M * 2 * 4);
    float* coeff = (float*)take((size_t)M * 8 * 4);
    float* gam   = (float*)take((size_t)M * 8 * 4);
    float* gb1   = (float*)take((size_t)E * I * 4);
    float* Ce    = (float*)take((size_t)E * D * 4);
    unsigned int* flags = (unsigned int*)take(16);

    // 1) x -> bf16
    k_cvt_bf16<<<(M * D + 255) / 256, 256, 0, stream>>>(x, xb, M * D);
    k_zero_flags<<<1, 32, 0, stream>>>(flags);

    // 2) router GEMM1: Hr = gelu(x @ rw1)   [M x 1024] bf16
    gemm_bf16_wmma<<<dim3(H2 / TN, M / TM), 256, 0, stream>>>(
        xb, D, rw1, H2, H2, D, nullptr, /*mode=*/0, Hbig, nullptr, nullptr, 0);

    // 3) router logits/softmax/top-2/active flags
    k_router_topk<<<M / 256, 256, 0, stream>>>(Hbig, rw2, wgt, idxA, flags);

    // 4) per-token expert coefficients + bias-leak gammas
    k_coeff_gamma<<<M / 256, 256, 0, stream>>>(wgt, idxA, flags, coeff, gam);

    // 5) Ce = gelu(b1[e]) @ w2[e] + b2[e]
    k_gelu_vec<<<(E * I + 255) / 256, 256, 0, stream>>>(b1, gb1, E * I);
    k_bias_leak<<<dim3(D / 256, E), 256, 0, stream>>>(gb1, w2, b2, Ce, D, I);

    // 6) out = Gamma @ Ce  (initializes output)
    k_init_out<<<(M * D) / 256, 256, 0, stream>>>(gam, Ce, out, M * D);

    // 7) dense expert passes: out += coeff[:,e] * ( gelu(x@W1e + b1e) @ W2e + b2e )
    for (int e = 0; e < E; ++e) {
        const float* w1e = w1 + (size_t)e * D * I;
        const float* w2e = w2 + (size_t)e * I * D;
        // H = gelu(x @ W1e + b1e)  -> bf16 [M x I]
        gemm_bf16_wmma<<<dim3(I / TN, M / TM), 256, 0, stream>>>(
            xb, D, w1e, I, I, D, b1 + (size_t)e * I, /*mode=*/0,
            Hbig, nullptr, nullptr, 0);
        // out += coeff[:,e] * (H @ W2e + b2e)
        gemm_bf16_wmma<<<dim3(D / TN, M / TM), 256, 0, stream>>>(
            Hbig, I, w2e, D, D, I, b2 + (size_t)e * D, /*mode=*/1,
            nullptr, out, coeff + e, 8);
    }
    (void)in_sizes; (void)n_in; (void)out_size; (void)ws_size;
}
